// RecurrentRetention_78082505441571
// MI455X (gfx1250) — matrix operand order repaired and checked by reading, not compiled, not run on hardware
//
#include <hip/hip_runtime.h>
#include <hip/hip_bf16.h>
#include <math.h>

#define BB 8
#define TT 2048
#define DD 1024
#define MM (BB * TT)            // 16384 rows of the flattened [B*T, D] GEMMs
#define GAMMA_VAL 0.96875f

typedef __bf16 bf16_t;
typedef __attribute__((ext_vector_type(16))) __bf16 v16bf;
typedef __attribute__((ext_vector_type(8)))  float  v8f;
typedef __attribute__((ext_vector_type(4)))  unsigned int v4u;
typedef __attribute__((ext_vector_type(4)))  unsigned int u32x4;
typedef __attribute__((ext_vector_type(8)))  int i32x8;
typedef __attribute__((ext_vector_type(4)))  int i32x4;

// ---- CDNA5 TDM availability (arity differs between toolchains; see bridge doc) ----
#if defined(__has_builtin)
#  if __has_builtin(__builtin_amdgcn_tensor_load_to_lds) && __has_builtin(__builtin_amdgcn_s_wait_tensorcnt)
#    define HAVE_TDM 1
#  endif
#endif
#if defined(__has_include)
#  if __has_include(<hip/amd_detail/amd_gfx1250_TDM.h>)
#    define TDM_6ARG 1
#  endif
#endif

union Frag16 { v16bf v; v4u q[2]; };

__device__ __forceinline__ v8f wmma_bf16(v16bf a, v16bf b, v8f c) {
    // D = A(16x32 bf16) * B(32x16 bf16) + C(16x16 f32)
    return __builtin_amdgcn_wmma_f32_16x16x32_bf16(
        /*neg_a=*/false, a, /*neg_b=*/false, b,
        /*c_mod=*/(short)0, c, /*reuse_a=*/false, /*reuse_b=*/false);
}

// A fragment, 16x32 (MxK), row-major A (global), leading dim lda.
__device__ __forceinline__ v16bf load_a_frag(const bf16_t* __restrict__ A, int lda,
                                             int m0, int k0, int lane) {
    const bf16_t* row = A + (size_t)(m0 + (lane & 15)) * lda;
    const int ks = (lane & 16) ? 8 : 0;
    Frag16 f;
    f.q[0] = *reinterpret_cast<const v4u*>(row + k0 + ks);
    f.q[1] = *reinterpret_cast<const v4u*>(row + k0 + 16 + ks);
    return f.v;
}

// A fragment sourced from an LDS-resident row-major tile (rows x 32 K-elements).
__device__ __forceinline__ v16bf lds_a_frag(const bf16_t* slice, int tile_i, int lane) {
    const bf16_t* row = slice + (tile_i * 16 + (lane & 15)) * 32 + ((lane & 16) ? 8 : 0);
    Frag16 f;
    f.q[0] = *reinterpret_cast<const v4u*>(row);
    f.q[1] = *reinterpret_cast<const v4u*>(row + 16);
    return f.v;
}

// B fragment, 32x16 (KxN), B given COLUMN-major as Bcm[n][k] (== W row-major
// for the x @ W^T einsum), leading dim ldb.
__device__ __forceinline__ v16bf load_b_frag(const bf16_t* __restrict__ Bcm, int ldb,
                                             int n0, int k0, int lane) {
    const bf16_t* p = Bcm + (size_t)(n0 + (lane & 15)) * ldb + k0 + ((lane & 16) ? 8 : 0);
    Frag16 f;
    f.q[0] = *reinterpret_cast<const v4u*>(p);
    f.q[1] = *reinterpret_cast<const v4u*>(p + 16);
    return f.v;
}

#if defined(HAVE_TDM)
// Issue one 2D Tensor-DMA: copy a (rows x 32) bf16 tile (row stride DD elems)
// from global into LDS at lds_addr, row-major packed. D# packed per ISA 8.3/8.4.
__device__ __forceinline__ void tdm_load_a_tile(unsigned lds_addr, const bf16_t* gptr, int rows) {
    unsigned long long ga = (unsigned long long)(uintptr_t)gptr;
    unsigned ga_lo = (unsigned)__builtin_amdgcn_readfirstlane((int)(unsigned)ga);
    unsigned ga_hi = (unsigned)__builtin_amdgcn_readfirstlane((int)(unsigned)(ga >> 32));
    unsigned lds_u = (unsigned)__builtin_amdgcn_readfirstlane((int)lds_addr);

    u32x4 g0;
    g0[0] = 1u;                                            // count=1, user descriptor
    g0[1] = lds_u;                                         // lds_addr
    g0[2] = ga_lo;                                         // global_addr[31:0]
    g0[3] = (ga_hi & 0x01FFFFFFu) | 0x80000000u;           // global_addr[56:32] | type=2

    // group1 bitfields (256b): data_size=1 (2B) @[17:16]; tensor_dim0=DD @[79:48];
    // tensor_dim1=MM @[111:80]; tile_dim0=32 @[127:112]; tile_dim1=rows @[143:128];
    // tensor_dim0_stride=DD @[207:160].
    unsigned long long qw0 = (1ull << 16) | ((unsigned long long)(DD & 0xFFFF) << 48);
    unsigned long long qw1 = ((unsigned long long)DD >> 16)
                           | ((unsigned long long)MM << 16)
                           | (32ull << 48);
    unsigned long long qw2 = (unsigned long long)(rows & 0xFFFF)
                           | ((unsigned long long)DD << 32);
    i32x8 g1;
    g1[0] = (int)(unsigned)qw0; g1[1] = (int)(unsigned)(qw0 >> 32);
    g1[2] = (int)(unsigned)qw1; g1[3] = (int)(unsigned)(qw1 >> 32);
    g1[4] = (int)(unsigned)qw2; g1[5] = (int)(unsigned)(qw2 >> 32);
    g1[6] = 0; g1[7] = 0;
    i32x4 gz = {0, 0, 0, 0};
#if defined(TDM_6ARG)
    i32x8 gz8 = {0, 0, 0, 0, 0, 0, 0, 0};
    __builtin_amdgcn_tensor_load_to_lds(g0, g1, gz, gz, gz8, 0);
#else
    __builtin_amdgcn_tensor_load_to_lds(g0, g1, gz, gz, 0);
#endif
}
#endif  // HAVE_TDM

// ---------------- fp32 -> bf16 conversion ----------------
__global__ void cvt_bf16_kernel(const float* __restrict__ src, bf16_t* __restrict__ dst, int n) {
    int i = blockIdx.x * blockDim.x + threadIdx.x;
    const int stride = gridDim.x * blockDim.x;
    for (; i < n; i += stride) dst[i] = (bf16_t)src[i];
}

// ---------------- u[t] = GAMMA * rowsum(W_ih[t,:]) + b_ih[t] ----------------
__global__ void u_bias_kernel(const float* __restrict__ Wih, const float* __restrict__ bih,
                              float* __restrict__ u) {
    __shared__ float red[256];
    const int t = blockIdx.x, tid = threadIdx.x;
    const float* row = Wih + (size_t)t * TT;
    float s = 0.f;
    for (int j = tid; j < TT; j += 256) s += row[j];
    red[tid] = s;
    __syncthreads();
    for (int off = 128; off > 0; off >>= 1) {
        if (tid < off) red[tid] += red[tid + off];
        __syncthreads();
    }
    if (tid == 0) u[t] = GAMMA_VAL * red[0] + bih[t];
}

__global__ void zero_kernel(float* __restrict__ p, int n) {
    int i = blockIdx.x * blockDim.x + threadIdx.x;
    if (i < n) p[i] = 0.f;
}

// ---------------- fused K/V GEMM + bias = rowsum(K .* V) ----------------
// grid (DD/32, MM/512), block 256 (8 waves); wave computes 64x32 of K AND of V
// via 4x2 tiles with dual accumulators. A-tiles (64x32 bf16) double-buffered in
// LDS by the Tensor Data Mover; K/V never touch memory.
__global__ __launch_bounds__(256) void kv_bias_gemm(const bf16_t* __restrict__ xb,
                                                    const bf16_t* __restrict__ wkb,
                                                    const bf16_t* __restrict__ wvb,
                                                    float* __restrict__ bias) {
    const int lane = threadIdx.x & 31;
    const int wave = threadIdx.x >> 5;
    const int m0 = blockIdx.y * 512 + wave * 64;
    const int n0 = blockIdx.x * 32;

    v8f ck[4][2] = {};
    v8f cv[4][2] = {};

#if defined(HAVE_TDM)
    extern __shared__ char smem_raw[];                  // 8 waves * 2 bufs * 4KB = 64KB
    bf16_t* smem = (bf16_t*)smem_raw;
    bf16_t* slice0 = smem + (wave * 2 + 0) * 2048;
    bf16_t* slice1 = smem + (wave * 2 + 1) * 2048;
    const unsigned lds0 = (unsigned)(uintptr_t)slice0;
    const unsigned lds1 = (unsigned)(uintptr_t)slice1;

    tdm_load_a_tile(lds0, xb + (size_t)m0 * DD, 64);    // prime buffer 0

    for (int k0 = 0; k0 < DD; k0 += 32) {
        const int buf = (k0 >> 5) & 1;
        const bf16_t* cur = buf ? slice1 : slice0;
        if (k0 + 32 < DD) {                             // async-load next tile, then
            tdm_load_a_tile(buf ? lds0 : lds1, xb + (size_t)m0 * DD + k0 + 32, 64);
            __builtin_amdgcn_s_wait_tensorcnt(1);       // wait: current tile resident
            __builtin_prefetch(wkb + (size_t)(n0 + (lane & 15)) * DD + k0 + 32, 0, 3);
        } else {
            __builtin_amdgcn_s_wait_tensorcnt(0);
        }
        v16bf bk0 = load_b_frag(wkb, DD, n0,      k0, lane);
        v16bf bk1 = load_b_frag(wkb, DD, n0 + 16, k0, lane);
        v16bf bv0 = load_b_frag(wvb, DD, n0,      k0, lane);
        v16bf bv1 = load_b_frag(wvb, DD, n0 + 16, k0, lane);
#pragma unroll
        for (int i = 0; i < 4; ++i) {
            v16bf a = lds_a_frag(cur, i, lane);
            ck[i][0] = wmma_bf16(a, bk0, ck[i][0]);
            ck[i][1] = wmma_bf16(a, bk1, ck[i][1]);
            cv[i][0] = wmma_bf16(a, bv0, cv[i][0]);
            cv[i][1] = wmma_bf16(a, bv1, cv[i][1]);
        }
    }
#else
    for (int k0 = 0; k0 < DD; k0 += 32) {
        v16bf bk0 = load_b_frag(wkb, DD, n0,      k0, lane);
        v16bf bk1 = load_b_frag(wkb, DD, n0 + 16, k0, lane);
        v16bf bv0 = load_b_frag(wvb, DD, n0,      k0, lane);
        v16bf bv1 = load_b_frag(wvb, DD, n0 + 16, k0, lane);
#pragma unroll
        for (int i = 0; i < 4; ++i) {
            v16bf a = load_a_frag(xb, DD, m0 + i * 16, k0, lane);
            ck[i][0] = wmma_bf16(a, bk0, ck[i][0]);
            ck[i][1] = wmma_bf16(a, bk1, ck[i][1]);
            cv[i][0] = wmma_bf16(a, bv0, cv[i][0]);
            cv[i][1] = wmma_bf16(a, bv1, cv[i][1]);
        }
    }
#endif

    // Per-row partial of sum_n K*V inside this wave's 32 N-columns.
    float p[4][8];
#pragma unroll
    for (int i = 0; i < 4; ++i)
#pragma unroll
        for (int v = 0; v < 8; ++v)
            p[i][v] = ck[i][0][v] * cv[i][0][v] + ck[i][1][v] * cv[i][1][v];
#pragma unroll
    for (int mk = 1; mk <= 8; mk <<= 1) {               // reduce over 16-lane N-groups
#pragma unroll
        for (int i = 0; i < 4; ++i)
#pragma unroll
            for (int v = 0; v < 8; ++v)
                p[i][v] += __shfl_xor(p[i][v], mk, 32);
    }
    if ((lane & 15) == 0) {                             // lanes 0 (M=v) and 16 (M=v+8)
        const int half8 = (lane >> 4) * 8;
#pragma unroll
        for (int i = 0; i < 4; ++i)
#pragma unroll
            for (int v = 0; v < 8; ++v)
                atomicAdd(&bias[m0 + i * 16 + half8 + v], p[i][v]);
    }
}

// ---------------- one tanh-RNN scan step ----------------
__global__ void rnn_step_kernel(const float* __restrict__ Whh, const float* __restrict__ u,
                                const float* __restrict__ bias_b,
                                const float* __restrict__ h_prev,
                                float* __restrict__ h_out, int first) {
    __shared__ float red[256];
    const int t = blockIdx.x, tid = threadIdx.x;
    float s = 0.f;
    if (!first) {
        const float* row = Whh + (size_t)t * TT;
#pragma unroll 4
        for (int j = tid; j < TT; j += 256) s += h_prev[j] * row[j];
    }
    red[tid] = s;
    __syncthreads();
    for (int off = 128; off > 0; off >>= 1) {
        if (tid < off) red[tid] += red[tid + off];
        __syncthreads();
    }
    if (tid == 0) h_out[t] = tanhf(u[t] + red[0] + bias_b[t]);
}

// ---------------- Q GEMM with fused output scaling ----------------
// grid (DD/32, MM/512), block 256 (8 waves); wave computes 64x32 via 4x2 tiles.
// A-tiles (64x32 bf16 = 4KB) are double-buffered in LDS by the Tensor Data Mover.
__global__ __launch_bounds__(256) void q_scale_gemm(const bf16_t* __restrict__ xb,
                                                    const bf16_t* __restrict__ wqb,
                                                    const float* __restrict__ S,
                                                    float* __restrict__ out) {
    const int lane = threadIdx.x & 31;
    const int wave = threadIdx.x >> 5;
    const int m0 = blockIdx.y * 512 + wave * 64;
    const int n0 = blockIdx.x * 32;

    v8f c[4][2] = {};

#if defined(HAVE_TDM)
    extern __shared__ char smem_raw[];                  // 8 waves * 2 bufs * 4KB = 64KB
    bf16_t* smem = (bf16_t*)smem_raw;
    bf16_t* slice0 = smem + (wave * 2 + 0) * 2048;
    bf16_t* slice1 = smem + (wave * 2 + 1) * 2048;
    const unsigned lds0 = (unsigned)(uintptr_t)slice0;
    const unsigned lds1 = (unsigned)(uintptr_t)slice1;

    tdm_load_a_tile(lds0, xb + (size_t)m0 * DD, 64);    // prime buffer 0

    for (int k0 = 0; k0 < DD; k0 += 32) {
        const int buf = (k0 >> 5) & 1;
        const bf16_t* cur = buf ? slice1 : slice0;
        if (k0 + 32 < DD) {                             // async-load next tile, then
            tdm_load_a_tile(buf ? lds0 : lds1, xb + (size_t)m0 * DD + k0 + 32, 64);
            __builtin_amdgcn_s_wait_tensorcnt(1);       // wait: current tile resident
        } else {
            __builtin_amdgcn_s_wait_tensorcnt(0);
        }
        v16bf b0 = load_b_frag(wqb, DD, n0,      k0, lane);
        v16bf b1 = load_b_frag(wqb, DD, n0 + 16, k0, lane);
#pragma unroll
        for (int i = 0; i < 4; ++i) {
            v16bf a = lds_a_frag(cur, i, lane);
            c[i][0] = wmma_bf16(a, b0, c[i][0]);
            c[i][1] = wmma_bf16(a, b1, c[i][1]);
        }
    }
#else
    for (int k0 = 0; k0 < DD; k0 += 32) {
        v16bf b0 = load_b_frag(wqb, DD, n0,      k0, lane);
        v16bf b1 = load_b_frag(wqb, DD, n0 + 16, k0, lane);
#pragma unroll
        for (int i = 0; i < 4; ++i) {
            v16bf a = load_a_frag(xb, DD, m0 + i * 16, k0, lane);
            c[i][0] = wmma_bf16(a, b0, c[i][0]);
            c[i][1] = wmma_bf16(a, b1, c[i][1]);
        }
    }
#endif

    const int col0 = n0 + (lane & 15);
    const int col1 = n0 + 16 + (lane & 15);
#pragma unroll
    for (int i = 0; i < 4; ++i) {
        const int rbase = m0 + i * 16 + ((lane >> 4) * 8);  // C layout: M = v + 8*(lane/16)
#pragma unroll
        for (int v = 0; v < 8; ++v) {
            const float s = S[rbase + v];
            out[(size_t)(rbase + v) * DD + col0] = s * c[i][0][v];
            out[(size_t)(rbase + v) * DD + col1] = s * c[i][1][v];
        }
    }
}

extern "C" void kernel_launch(void* const* d_in, const int* in_sizes, int n_in,
                              void* d_out, int out_size, void* d_ws, size_t ws_size,
                              hipStream_t stream) {
    const float* x   = (const float*)d_in[0];
    const float* Wq  = (const float*)d_in[1];
    const float* Wk  = (const float*)d_in[2];
    const float* Wv  = (const float*)d_in[3];
    const float* Wih = (const float*)d_in[4];
    const float* Whh = (const float*)d_in[5];
    const float* bih = (const float*)d_in[6];
    float* out = (float*)d_out;

    // workspace layout (~40 MB): xb | wqb | wkb | wvb | u | bias | S
    char* ws = (char*)d_ws;
    size_t off = 0;
    bf16_t* xb  = (bf16_t*)(ws + off); off += (size_t)MM * DD * sizeof(bf16_t);
    bf16_t* wqb = (bf16_t*)(ws + off); off += (size_t)DD * DD * sizeof(bf16_t);
    bf16_t* wkb = (bf16_t*)(ws + off); off += (size_t)DD * DD * sizeof(bf16_t);
    bf16_t* wvb = (bf16_t*)(ws + off); off += (size_t)DD * DD * sizeof(bf16_t);
    float*  u    = (float*)(ws + off); off += (size_t)TT * sizeof(float);
    float*  bias = (float*)(ws + off); off += (size_t)MM * sizeof(float);
    float*  S    = (float*)(ws + off); off += (size_t)MM * sizeof(float);

#if defined(HAVE_TDM)
    const size_t shbytes = 8 * 2 * 64 * 32 * sizeof(bf16_t);  // 64 KB
#else
    const size_t shbytes = 0;
#endif

    // 1) precision conversion
    cvt_bf16_kernel<<<8192, 256, 0, stream>>>(x,  xb,  MM * DD);
    cvt_bf16_kernel<<<1024, 256, 0, stream>>>(Wq, wqb, DD * DD);
    cvt_bf16_kernel<<<1024, 256, 0, stream>>>(Wk, wkb, DD * DD);
    cvt_bf16_kernel<<<1024, 256, 0, stream>>>(Wv, wvb, DD * DD);

    // 2) u vector and bias accumulator init
    u_bias_kernel<<<TT, 256, 0, stream>>>(Wih, bih, u);
    zero_kernel<<<MM / 256, 256, 0, stream>>>(bias, MM);

    // 3) fused K/V WMMA GEMM (TDM-staged A tiles) -> bias (K, V never materialized)
    kv_bias_gemm<<<dim3(DD / 32, MM / 512), 256, shbytes, stream>>>(xb, wkb, wvb, bias);

    // 4) tanh-RNN scan over the B dimension (8 serial GEMV steps; W_hh is L2-resident)
    for (int b = 0; b < BB; ++b) {
        const float* h_prev = (b == 0) ? S : (S + (size_t)(b - 1) * TT);
        rnn_step_kernel<<<TT, 256, 0, stream>>>(Whh, u, bias + (size_t)b * TT,
                                                h_prev, S + (size_t)b * TT, b == 0);
    }

    // 5) Q WMMA GEMM (TDM double-buffered A tiles) with fused S-scaling into d_out
    q_scale_gemm<<<dim3(DD / 32, MM / 512), 256, shbytes, stream>>>(xb, wqb, S, out);
}
